// NeuralHawkesModel_36816459661476
// MI455X (gfx1250) — matrix operand order repaired and checked by reading, not compiled, not run on hardware
//
#include <hip/hip_runtime.h>
#include <hip/hip_bf16.h>
#include <math.h>

// Problem constants (Neural Hawkes CT-LSTM)
#define BATCH 64
#define SEQ   512
#define EMB   256
#define HID   512
#define NK    100
#define NKP   112      // NK padded to 7*16 for WMMA N tiles
#define EH    768      // EMB + HID
#define FH    2560     // 5 * HID

// LDS layout for nh_scan (dynamic, 287296 B < 320 KB WGP LDS)
#define BLD   776      // W slice column stride: 388 dw == 4 (mod 64) -> conflict-free
#define XLD   264      // X tile row stride: 132 dw == 4 (mod 64)  -> conflict-free
#define HLD   552      // H tile row stride: 276 dw == 20 (mod 64) -> conflict-free
#define OFF_B   0                        // 160*776*2 = 248320
#define OFF_X   248320                   // 16*264*2  = 8448
#define OFF_H   (248320 + 8448)          // 16*552*2  = 17664
#define OFF_Z   (OFF_H + 17664)          // 16*168*4  = 10752
#define OFF_C   (OFF_Z + 10752)          // 16*32*4   = 2048
#define OFF_EV  (OFF_C + 2048)
#define SMEM_SCAN (OFF_EV + 64)          // 287296

// ---- CDNA5 data-mover availability (guarded: falls back to sync copies) ---
#if defined(__gfx1250__) && __has_builtin(__builtin_amdgcn_global_load_async_to_lds_b128)
#define NH_ASYNC 1
#else
#define NH_ASYNC 0
#endif
#if defined(__gfx1250__) && __has_builtin(__builtin_amdgcn_tensor_load_to_lds) && \
    __has_builtin(__builtin_amdgcn_s_wait_tensorcnt)
#define NH_TDM 1
#else
#define NH_TDM 0
#endif

typedef __attribute__((ext_vector_type(16))) __bf16 bf16x16;
typedef __attribute__((ext_vector_type(8)))  __bf16 bf16x8;
typedef __attribute__((ext_vector_type(8)))  float  floatx8;
typedef __attribute__((ext_vector_type(4))) unsigned uint32x4;
typedef __attribute__((ext_vector_type(8))) int      int32x8;
typedef __attribute__((ext_vector_type(4))) int      int32x4;

#if NH_ASYNC
typedef __attribute__((address_space(1))) int32x4 gas_i32x4;  // global v4i32
typedef __attribute__((address_space(3))) int32x4 las_i32x4;  // LDS v4i32
#endif

// ---- scalar helpers -------------------------------------------------------
__device__ inline __bf16 f2bf(float f) {
  unsigned u = __float_as_uint(f);
  u += 0x7FFFu + ((u >> 16) & 1u);           // round-to-nearest-even
  unsigned short s = (unsigned short)(u >> 16);
  __bf16 r; __builtin_memcpy(&r, &s, 2); return r;
}
__device__ inline float sigmoidf_(float x) { return 1.f / (1.f + expf(-x)); }
__device__ inline float softplusf_(float x) { return (x > 20.f) ? x : log1pf(expf(x)); }

// ---- 16-byte global->LDS copy: async on CDNA5, sync fallback --------------
__device__ inline void copy16(__bf16* lds_dst, const __bf16* gsrc) {
#if NH_ASYNC
  __builtin_amdgcn_global_load_async_to_lds_b128(
      (gas_i32x4*)(size_t)(const void*)gsrc,
      (las_i32x4*)(unsigned)(size_t)(void*)lds_dst,
      0, 0);
#else
  *(bf16x8*)lds_dst = *(const bf16x8*)gsrc;
#endif
}
__device__ inline void copy16_wait() {
#if NH_ASYNC
  asm volatile("s_wait_asynccnt 0x0" ::: "memory");
#endif
}

// ---- TDM: load 16x512 bf16 h row-block into H tile with LDS padding -------
// tile_dim0=512 elems (1024B rows), tile_dim1=16 rows, tensor_dim0_stride=512.
// pad_enable=1, pad_interval=7 (pad after every 128*8B = one row),
// pad_amount=19 (+20 dwords = 80B) -> LDS row stride 1104B = HLD elems.
#if NH_TDM
__device__ inline void tdm_load_h(const __bf16* hsrc, void* lds_dst) {
  size_t   ga  = (size_t)(const void*)hsrc;
  unsigned lds = (unsigned)(size_t)lds_dst;
  uint32x4 g0;
  g0[0] = 1u;                                   // count=1, user descriptor
  g0[1] = lds;                                  // lds_addr
  g0[2] = (unsigned)ga;                         // global_addr[31:0]
  g0[3] = (unsigned)((ga >> 32) & 0x01FFFFFFu) | 0x80000000u;  // GA[56:32]|type=2
  int32x8 g1;
  g1[0] = (int)((1u << 16) | (1u << 20) | (7u << 22) | (19u << 25));
  g1[1] = (int)(512u << 16);                    // tensor_dim0 = 512
  g1[2] = (int)(16u << 16);                     // tensor_dim1 = 16
  g1[3] = (int)(512u << 16);                    // tile_dim0 = 512
  g1[4] = 16;                                   // tile_dim1 = 16
  g1[5] = 512;                                  // tensor_dim0_stride = 512
  g1[6] = 0; g1[7] = 0;
  int32x4 gz; gz[0] = gz[1] = gz[2] = gz[3] = 0;
#if __clang_major__ >= 23
  int32x8 z8; z8[0]=z8[1]=z8[2]=z8[3]=z8[4]=z8[5]=z8[6]=z8[7]=0;
  __builtin_amdgcn_tensor_load_to_lds(g0, g1, gz, gz, z8, 0);
#else
  __builtin_amdgcn_tensor_load_to_lds(g0, g1, gz, gz, 0);
#endif
}
#endif

// ---- WMMA helpers (layouts per CDNA5 ISA 7.12.2, wave32) ------------------
__device__ inline floatx8 wmma_bf16(bf16x16 a, bf16x16 b, floatx8 c) {
  return __builtin_amdgcn_wmma_f32_16x16x32_bf16(false, a, false, b, (short)0, c,
                                                 false, false);
}
__device__ inline bf16x16 cat8(bf16x8 lo, bf16x8 hi) {
  return __builtin_shufflevector(lo, hi, 0, 1, 2, 3, 4, 5, 6, 7,
                                 8, 9, 10, 11, 12, 13, 14, 15);
}
// B-matrix 32x16 bf16 stored K-MAJOR (column n contiguous in K, stride ld).
__device__ inline bf16x16 load_b_kmajor(const __bf16* Bc, int ld, int k0, int lane) {
  int n = lane & 15, kq = (lane >> 4) * 8;
  const __bf16* p = Bc + (size_t)n * ld + k0 + kq;
  return cat8(*(const bf16x8*)p, *(const bf16x8*)(p + 16));
}
// A-matrix 16x32 bf16 (row-major). lane<16: row=lane, K={0..7,16..23};
// lane>=16: K={8..15,24..31}.
__device__ inline bf16x16 load_a_row(const __bf16* Arow0, int lda, int k, int lane) {
  int row = lane & 15, kh = (lane >> 4) * 8;
  const __bf16* p = Arow0 + row * lda + k + kh;
  return cat8(*(const bf16x8*)p, *(const bf16x8*)(p + 16));
}

// ---- Phase 0: convert + TRANSPOSE weights to K-major bf16 -----------------
__global__ void nh_prep(const float* __restrict__ emb, const float* __restrict__ W,
                        const float* __restrict__ Wi,
                        __bf16* __restrict__ emb_bf, __bf16* __restrict__ W_bfT,
                        __bf16* __restrict__ Wi_bfT) {
  const int NE  = NK * EMB;       // 25600
  const int NW  = EH * FH;        // 1966080
  const int NWI = NKP * HID;      // 57344
  int i = blockIdx.x * blockDim.x + threadIdx.x;
  if (i < NE) emb_bf[i] = f2bf(emb[i]);
  int j = i - NE;
  if (j >= 0 && j < NW) {         // W_bfT[n][k] = W[k][n]
    int n = j / EH, kk = j - n * EH;
    W_bfT[j] = f2bf(W[(size_t)kk * FH + n]);
  }
  int k = i - NE - NW;
  if (k >= 0 && k < NWI) {        // Wi_bfT[n][k] = Wi[k][n], zero-pad n>=100
    int n = k / HID, kk = k - n * HID;
    Wi_bfT[k] = (n < NK) ? f2bf(Wi[(size_t)kk * NK + n]) : f2bf(0.f);
  }
}

// ---- Phase 1: recurrent scan ---------------------------------------------
// 64 persistent WGs: 4 batch tiles (16 rows = WMMA M) x clusters of 16 WGs.
// Each WG owns 32 hidden units (5 gates -> 160 W columns = 10 tiles); its
// whole W slice lives in LDS (~243 KB) for all 512 steps (CDNA5 320 KB LDS).
__global__ __launch_bounds__(320)
void nh_scan(const int* __restrict__ evt, const float* __restrict__ dt,
             const float* __restrict__ bias, const __bf16* __restrict__ emb_bf,
             const __bf16* __restrict__ W_bfT, __bf16* h_buf, unsigned* ctr,
             float* __restrict__ hid_out, float* __restrict__ dec_out) {
  extern __shared__ char smem[];
  __bf16* Bbuf   = (__bf16*)(smem + OFF_B);   // [160][BLD] K-major W slice
  __bf16* Xtile  = (__bf16*)(smem + OFF_X);   // [16][XLD]  embedding part of A
  __bf16* Htile  = (__bf16*)(smem + OFF_H);   // [16][HLD]  h part of A
  float*  zbuf   = (float*) (smem + OFF_Z);   // [16][168]  gate pre-activations
  float*  cstate = (float*) (smem + OFF_C);   // [16][32]   persistent cell state
  int*    evs    = (int*)   (smem + OFF_EV);

  const int tid  = threadIdx.x;
  const int lane = tid & 31;
  const int wave = tid >> 5;                  // 0..9 -> output tile id
  const int bt = blockIdx.x >> 4;             // batch tile 0..3
  const int wg = blockIdx.x & 15;             // hidden slice 0..15
  const int b0 = bt * 16;
  const int u0 = wg * 32;

  // one-time: stage this WG's 160-column W slice into LDS (K-major, 16B async)
  for (int q = tid; q < 160 * (EH / 8); q += 320) {
    int col = q / (EH / 8), koff = (q % (EH / 8)) * 8;
    int g = col >> 5, j = col & 31;
    copy16(Bbuf + (size_t)col * BLD + koff,
           W_bfT + (size_t)(g * HID + u0 + j) * EH + koff);
  }
  for (int it = tid; it < 16 * 32; it += 320) cstate[it] = 0.f;
  copy16_wait();
  __syncthreads();

  const int g_t = wave >> 1, jb_t = (wave & 1) * 16;   // tile -> gate/sub-col
  const __bf16* Btile = Bbuf + (size_t)(wave * 16) * BLD;
  const float*  biasp = bias + g_t * HID + u0 + jb_t;

  for (int s = 0; s < SEQ; ++s) {
    const int cur = s & 1, nxt = cur ^ 1;
    const __bf16* hcur = h_buf + (size_t)cur * (BATCH * HID);

    if (tid < 16) evs[tid] = evt[(b0 + tid) * SEQ + s];
#if NH_TDM
    if (wave == 0) tdm_load_h(hcur + (size_t)b0 * HID, (void*)Htile);
#endif
    __syncthreads();

    // stage x = embedding[ev] (16B async chunks)
    for (int q = tid; q < 16 * (EMB / 8); q += 320) {
      int r = q / (EMB / 8), c = (q % (EMB / 8)) * 8;
      copy16(Xtile + r * XLD + c, emb_bf + (size_t)evs[r] * EMB + c);
    }
#if !NH_TDM
    for (int q = tid; q < 16 * (HID / 8); q += 320) {
      int r = q >> 6, c = (q & 63) * 8;
      copy16(Htile + r * HLD + c, hcur + (size_t)(b0 + r) * HID + c);
    }
#endif
    copy16_wait();
#if NH_TDM
    if (wave == 0) __builtin_amdgcn_s_wait_tensorcnt(0);
#endif
    __syncthreads();

    // [16 x 768] @ [768 x 16] per wave, all operands LDS-resident
    floatx8 acc = {};
#pragma unroll
    for (int k0 = 0; k0 < EH; k0 += 32) {
      bf16x16 av = (k0 < EMB) ? load_a_row(Xtile, XLD, k0, lane)
                              : load_a_row(Htile, HLD, k0 - EMB, lane);
      bf16x16 bv = load_b_kmajor(Btile, BLD, k0, lane);
      acc = wmma_bf16(av, bv, acc);
    }
    {
      int n = lane & 15;
      float bv = biasp[n];
      int rbase = (lane >> 4) * 8;              // C/D: row = v + 8*(lane>>4)
#pragma unroll
      for (int v = 0; v < 8; ++v)
        zbuf[(rbase + v) * 168 + wave * 16 + n] = acc[v] + bv;
    }
    __syncthreads();

    // gates + continuous-time cell update (16 rows x 32 units)
    for (int it = tid; it < 16 * 32; it += 320) {
      int r = it >> 5, j = it & 31;
      const float* zr = zbuf + r * 168;
      float ig  = sigmoidf_(zr[0 * 32 + j]);
      float fg  = sigmoidf_(zr[1 * 32 + j]);
      float og  = sigmoidf_(zr[2 * 32 + j]);
      float gg  = tanhf   (zr[3 * 32 + j]);
      float dec = softplusf_(zr[4 * 32 + j]);
      float dts = dt[(b0 + r) * SEQ + s];
      float cn  = fg * cstate[it] + ig * gg;
      float ct  = cn * expf(-dec * dts);
      float h   = og * tanhf(ct);
      cstate[it] = ct;
      size_t o = ((size_t)(b0 + r) * SEQ + s) * HID + u0 + j;
      hid_out[o] = h;
      dec_out[o] = dec;
      h_buf[(size_t)nxt * (BATCH * HID) + (b0 + r) * HID + u0 + j] = f2bf(h);
    }

    // ---- inter-workgroup step barrier (16 WGs per batch tile) ----
    __threadfence();
    __syncthreads();
    __builtin_amdgcn_s_cluster_barrier();  // s_barrier_signal/wait -3 (NOP unclustered)
    if (tid == 0) {
      atomicAdd(&ctr[bt], 1u);
      volatile unsigned* vc = ctr + bt;
      unsigned target = 16u * (unsigned)(s + 1);
      while (*vc < target) __builtin_amdgcn_s_sleep(2);
    }
    __syncthreads();
    __threadfence();
  }
}

// ---- Phase 2: intensities = softplus(h_seq @ Wi + bi) ---------------------
__global__ __launch_bounds__(256)
void nh_intensity(const float* __restrict__ hid, const __bf16* __restrict__ Wi_bfT,
                  const float* __restrict__ bi, float* __restrict__ inten) {
  __shared__ __bf16 At[16 * 520];
  const int tid = threadIdx.x, lane = tid & 31, wave = tid >> 5;
  const int m0 = blockIdx.x * 16;

  for (int q = tid; q < 16 * (HID / 8); q += 256) {   // f32 -> bf16, 16B stores
    int r = q >> 6, c = (q & 63) * 8;
    const float* src = hid + (size_t)(m0 + r) * HID + c;
    bf16x8 v;
#pragma unroll
    for (int i = 0; i < 8; ++i) v[i] = f2bf(src[i]);
    *(bf16x8*)(At + r * 520 + c) = v;
  }
  __syncthreads();

  if (wave < 7) {
    const int n0 = wave * 16;
    floatx8 acc = {};
#pragma unroll
    for (int k0 = 0; k0 < HID; k0 += 32) {
      bf16x16 av = load_a_row(At, 520, k0, lane);
      bf16x16 bv = load_b_kmajor(Wi_bfT + (size_t)n0 * HID, HID, k0, lane);
      acc = wmma_bf16(av, bv, acc);
    }
    int n = lane & 15, col = n0 + n;
    if (col < NK) {
      float bb = bi[col];
      int rbase = (lane >> 4) * 8;
#pragma unroll
      for (int v = 0; v < 8; ++v) {
        float x = acc[v] + bb;
        inten[(size_t)(m0 + rbase + v) * NK + col] = softplusf_(x);
      }
    }
  }
}

// ---- host side ------------------------------------------------------------
extern "C" void kernel_launch(void* const* d_in, const int* in_sizes, int n_in,
                              void* d_out, int out_size, void* d_ws, size_t ws_size,
                              hipStream_t stream) {
  (void)in_sizes; (void)n_in; (void)out_size; (void)ws_size;
  const int*   evt  = (const int*)  d_in[0];
  const float* dt   = (const float*)d_in[1];
  const float* emb  = (const float*)d_in[2];
  const float* W    = (const float*)d_in[3];
  const float* bias = (const float*)d_in[4];
  const float* Wi   = (const float*)d_in[5];
  const float* bi   = (const float*)d_in[6];

  char* ws = (char*)d_ws;
  size_t off = 0;
  auto alloc = [&](size_t bytes) -> void* {
    void* p = ws + off;
    off = (off + bytes + 255) & ~(size_t)255;
    return p;
  };
  __bf16*   emb_bf = (__bf16*)alloc((size_t)NK * EMB * 2);
  __bf16*   W_bfT  = (__bf16*)alloc((size_t)EH * FH * 2);
  __bf16*   Wi_bfT = (__bf16*)alloc((size_t)NKP * HID * 2);
  __bf16*   h_buf  = (__bf16*)alloc((size_t)2 * BATCH * HID * 2);
  unsigned* ctr    = (unsigned*)alloc(256);

  float* hid_out = (float*)d_out;
  float* dec_out = hid_out + (size_t)BATCH * SEQ * HID;
  float* inten   = dec_out + (size_t)BATCH * SEQ * HID;

  // h0 = 0 and barrier counters = 0 (graph-capture safe, replayed each call)
  (void)hipMemsetAsync(h_buf, 0, (size_t)2 * BATCH * HID * 2, stream);
  (void)hipMemsetAsync(ctr, 0, 256, stream);

  // allow ~287 KB dynamic LDS (CDNA5 WGP: up to 320 KB per workgroup)
  (void)hipFuncSetAttribute((const void*)nh_scan,
                            hipFuncAttributeMaxDynamicSharedMemorySize, SMEM_SCAN);

  const int prep_total = NK * EMB + EH * FH + NKP * HID;
  nh_prep<<<(prep_total + 255) / 256, 256, 0, stream>>>(emb, W, Wi,
                                                        emb_bf, W_bfT, Wi_bfT);
  nh_scan<<<64, 320, SMEM_SCAN, stream>>>(evt, dt, bias, emb_bf, W_bfT, h_buf, ctr,
                                          hid_out, dec_out);
  nh_intensity<<<(BATCH * SEQ) / 16, 256, 0, stream>>>(hid_out, Wi_bfT, bi, inten);
}
